// MAFS_82824149336585
// MI455X (gfx1250) — matrix operand order repaired and checked by reading, not compiled, not run on hardware
//
#include <hip/hip_runtime.h>

typedef __attribute__((ext_vector_type(16))) __bf16 v16bf;
typedef __attribute__((ext_vector_type(8)))  float  v8f;
typedef __bf16 bf16;

#define B_   2
#define Q_   900
#define C_   256
#define NCAM 6
#define HF   112
#define WF   200
#define ZL   10
#define YL   128
#define XL   128
#define MTOT (B_ * Q_)   /* 1800 */
#define EPS_ 1e-5f

// ---------------------------------------------------------------------------
// Kernel 0: convert all weight matrices to bf16, TRANSPOSED to N x K layout so
// a WMMA B-fragment (col n, K = h*8+{0..7} and 16+h*8+{0..7}) is two 16B loads.
// ---------------------------------------------------------------------------
__global__ __launch_bounds__(256, 1)
void k_cvt_weights(const float* __restrict__ cam_w,
                   const float* __restrict__ lidar_w,
                   const float* __restrict__ fus_w1,
                   const float* __restrict__ fus_w2,
                   bf16* __restrict__ cwT, bf16* __restrict__ lwT,
                   bf16* __restrict__ f1T, bf16* __restrict__ f2T) {
  const int i = blockIdx.x * blockDim.x + threadIdx.x;
  if (i < 256 * 256) {
    const int n = i >> 8, k = i & 255;          // out[n*256+k] = in[k*256+n]
    cwT[i] = (bf16)cam_w[k * 256 + n];
    lwT[i] = (bf16)lidar_w[k * 256 + n];
    f2T[i] = (bf16)fus_w2[k * 256 + n];
  }
  if (i < 512 * 256) {
    const int n = i >> 9, k = i & 511;          // fus_w1 is (512,256): K=512,N=256
    f1T[i] = (bf16)fus_w1[k * 256 + n];
  }
}

// ---------------------------------------------------------------------------
// Kernel 1: camera projection + bilinear sampling (masked mean over 6 cams),
// LiDAR trilinear sampling, and the gating MLP + softmax.
// One block per (b,q); one thread per channel.
// ---------------------------------------------------------------------------
__global__ __launch_bounds__(256, 1)
void k_sample_gate(const float* __restrict__ query,
                   const float* __restrict__ refp,
                   const float* __restrict__ img,
                   const float* __restrict__ lidar,
                   const float* __restrict__ l2i,
                   const float* __restrict__ att_w1,
                   const float* __restrict__ att_b1,
                   const float* __restrict__ att_w2,
                   const float* __restrict__ att_b2,
                   float* __restrict__ feat_cam,
                   float* __restrict__ feat_lid,
                   float* __restrict__ gate) {
  __shared__ float sq[C_];
  __shared__ float sh[64];
  __shared__ float slog[2];
  const int bq  = blockIdx.x;
  const int b   = bq / Q_;
  const int tid = threadIdx.x;

  sq[tid] = query[bq * C_ + tid];

  const float rx = refp[bq * 3 + 0], ry = refp[bq * 3 + 1], rz = refp[bq * 3 + 2];
  const float ax = rx * 102.4f - 51.2f;   // PC_MIN/PC_MAX de-normalization
  const float ay = ry * 102.4f - 51.2f;
  const float az = rz * 8.0f   - 5.0f;

  // ---- camera branch ----
  float acc = 0.f, cnt = 0.f;
  for (int n = 0; n < NCAM; ++n) {
    const float* M = l2i + (b * NCAM + n) * 16;
    const float c0 = M[0] * ax + M[1] * ay + M[2]  * az + M[3];
    const float c1 = M[4] * ax + M[5] * ay + M[6]  * az + M[7];
    const float c2 = M[8] * ax + M[9] * ay + M[10] * az + M[11];
    const float d  = fmaxf(c2, EPS_);
    const float u  = 2.f * (c0 / d) * (1.f / 1599.f) - 1.f;   // W0-1
    const float v  = 2.f * (c1 / d) * (1.f / 899.f)  - 1.f;   // H0-1
    const bool valid = (u >= -1.f) && (u <= 1.f) && (v >= -1.f) && (v <= 1.f) && (c2 > EPS_);
    if (!valid) continue;
    cnt += 1.f;
    const float x = ((u + 1.f) * WF - 1.f) * 0.5f;
    const float y = ((v + 1.f) * HF - 1.f) * 0.5f;
    const float xf = floorf(x), yf = floorf(y);
    const float fx = x - xf, fy = y - yf;
    const int ix = (int)xf, iy = (int)yf;
    const float wx[2] = {1.f - fx, fx};
    const float wy[2] = {1.f - fy, fy};
    const float* base = img + ((size_t)((b * NCAM + n) * C_ + tid)) * (size_t)(HF * WF);
    float s = 0.f;
#pragma unroll
    for (int dy = 0; dy < 2; ++dy) {
#pragma unroll
      for (int dx = 0; dx < 2; ++dx) {
        const int xi = ix + dx, yi = iy + dy;
        if (xi >= 0 && xi < WF && yi >= 0 && yi < HF)
          s += wx[dx] * wy[dy] * base[yi * WF + xi];
      }
    }
    acc += s;
  }
  feat_cam[bq * C_ + tid] = acc / fmaxf(cnt, 1.f);

  // ---- LiDAR branch (trilinear, zero padding) ----
  {
    const float gx = rx * 2.f - 1.f, gy = ry * 2.f - 1.f, gz = rz * 2.f - 1.f;
    const float x = ((gx + 1.f) * XL - 1.f) * 0.5f;
    const float y = ((gy + 1.f) * YL - 1.f) * 0.5f;
    const float z = ((gz + 1.f) * ZL - 1.f) * 0.5f;
    const float xf = floorf(x), yf = floorf(y), zf = floorf(z);
    const int ix = (int)xf, iy = (int)yf, iz = (int)zf;
    const float wx[2] = {1.f - (x - xf), x - xf};
    const float wy[2] = {1.f - (y - yf), y - yf};
    const float wz[2] = {1.f - (z - zf), z - zf};
    const float* base = lidar + ((size_t)(b * C_ + tid)) * (size_t)(ZL * YL * XL);
    float s = 0.f;
#pragma unroll
    for (int dz = 0; dz < 2; ++dz)
#pragma unroll
      for (int dy = 0; dy < 2; ++dy)
#pragma unroll
        for (int dx = 0; dx < 2; ++dx) {
          const int xi = ix + dx, yi = iy + dy, zi = iz + dz;
          if (xi >= 0 && xi < XL && yi >= 0 && yi < YL && zi >= 0 && zi < ZL)
            s += wx[dx] * wy[dy] * wz[dz] * base[(zi * YL + yi) * XL + xi];
        }
    feat_lid[bq * C_ + tid] = s;
  }

  // ---- gating MLP: relu(q@W1+b1) @ W2 + b2 -> softmax(2) ----
  __syncthreads();
  if (tid < 64) {
    float h = att_b1[tid];
    for (int k = 0; k < C_; ++k) h += sq[k] * att_w1[k * 64 + tid];
    sh[tid] = fmaxf(h, 0.f);
  }
  __syncthreads();
  if (tid < 2) {
    float l = att_b2[tid];
    for (int j = 0; j < 64; ++j) l += sh[j] * att_w2[j * 2 + tid];
    slog[tid] = l;
  }
  __syncthreads();
  if (tid == 0) {
    const float m  = fmaxf(slog[0], slog[1]);
    const float e0 = __expf(slog[0] - m), e1 = __expf(slog[1] - m);
    const float inv = 1.f / (e0 + e1);
    gate[bq * 2 + 0] = e0 * inv;
    gate[bq * 2 + 1] = e1 * inv;
  }
}

// ---------------------------------------------------------------------------
// WMMA 16x16 tile accumulation over K (bf16 in, f32 accumulate).
// A staged in LDS (16 rows, row stride K): lane (h,m) needs K = h*8+{0..7} and
// 16+h*8+{0..7} per 32-K step (two contiguous 16B runs, per ISA A layout).
// B is bf16 transposed N x K: lane (h,n) needs K = h*8+{0..7} and 16+h*8+{0..7}.
// ---------------------------------------------------------------------------
template <int K>
__device__ __forceinline__ v8f wmma_tile(const bf16* ldsA,
                                         const bf16* __restrict__ Bt,
                                         int ncol0, int lane, v8f acc) {
  const int h = lane >> 4, mn = lane & 15;
  const bf16* arow = ldsA + mn * K;
  const bf16* brow = Bt + (size_t)(ncol0 + mn) * K;
#pragma unroll
  for (int k0 = 0; k0 < K; k0 += 32) {
    v16bf a, bb;
    const bf16* a0 = arow + k0 + h * 8;
    const bf16* b0 = brow + k0 + h * 8;
#pragma unroll
    for (int e = 0; e < 8; ++e) { a[e] = a0[e]; a[e + 8] = a0[16 + e]; }
#pragma unroll
    for (int e = 0; e < 8; ++e) { bb[e] = b0[e]; bb[e + 8] = b0[16 + e]; }
    acc = __builtin_amdgcn_wmma_f32_16x16x32_bf16(false, a, false, bb,
                                                  (short)0, acc, false, false);
  }
  return acc;
}

// ---------------------------------------------------------------------------
// Kernel 2: cam/lidar projections (two 256x256 GEMMs) + gated bf16 concat.
// Block = 16-row M-tile, 8 waves; two sequential phases (cam, then lidar) to
// keep only one accumulator chain live; wave handles N-tiles {w, w+8}.
// ---------------------------------------------------------------------------
__global__ __launch_bounds__(256, 1)
void k_proj_fuse(const float* __restrict__ feat_cam,
                 const float* __restrict__ feat_lid,
                 const bf16* __restrict__ cwT,
                 const bf16* __restrict__ lwT,
                 const float* __restrict__ cam_b,
                 const float* __restrict__ lidar_b,
                 const float* __restrict__ gate,
                 bf16* __restrict__ fused) {
  __shared__ bf16 Ac[16 * 256];
  __shared__ bf16 Al[16 * 256];
  const int row0 = blockIdx.x * 16;
  const int tid  = threadIdx.x;
  for (int i = tid; i < 16 * 256; i += 256) {
    const int r = i >> 8, k = i & 255;
    const int row = row0 + r;
    float vc = 0.f, vl = 0.f;
    if (row < MTOT) { vc = feat_cam[row * 256 + k]; vl = feat_lid[row * 256 + k]; }
    Ac[i] = (bf16)vc; Al[i] = (bf16)vl;
  }
  __syncthreads();
  const int wave = tid >> 5, lane = tid & 31;
  const int h = lane >> 4, nn = lane & 15;
#pragma clang loop unroll(disable)
  for (int s = 0; s < 2; ++s) {                 // phase: 0 = camera, 1 = lidar
    const bf16*  Asrc = s ? Al : Ac;
    const bf16*  Wt   = s ? lwT : cwT;
    const float* bias = s ? lidar_b : cam_b;
#pragma clang loop unroll(disable)
    for (int t = 0; t < 2; ++t) {
      const int nt  = wave + t * 8;
      const int col = nt * 16 + nn;
      v8f z = {};
      v8f acc = wmma_tile<256>(Asrc, Wt, nt * 16, lane, z);
      const float bv = bias[col];
#pragma unroll
      for (int r = 0; r < 8; ++r) {
        const int row = row0 + r + h * 8;
        if (row < MTOT) {
          const float g = gate[row * 2 + s];
          fused[(size_t)row * 512 + s * 256 + col] = (bf16)(g * (acc[r] + bv));
        }
      }
    }
  }
}

// ---------------------------------------------------------------------------
// Kernel 3: fused @ fus_w1 + b1 (K=512) -> LayerNorm -> ReLU -> bf16.
// ---------------------------------------------------------------------------
__global__ __launch_bounds__(256, 1)
void k_fus1_ln(const bf16* __restrict__ fused,
               const bf16* __restrict__ f1T,
               const float* __restrict__ b1,
               const float* __restrict__ ln_g,
               const float* __restrict__ ln_b,
               bf16* __restrict__ h2) {
  __shared__ bf16  A[16 * 512];
  __shared__ float Cb[16 * 256];
  __shared__ float mu_[16], rs_[16];
  const int row0 = blockIdx.x * 16;
  const int tid  = threadIdx.x;
  for (int i = tid; i < 16 * 512; i += 256) {
    const int r = i >> 9, k = i & 511;
    const int row = row0 + r;
    A[i] = (row < MTOT) ? fused[(size_t)row * 512 + k] : (bf16)0.f;
  }
  __syncthreads();
  const int wave = tid >> 5, lane = tid & 31;
  const int h = lane >> 4, nn = lane & 15;
#pragma clang loop unroll(disable)
  for (int t = 0; t < 2; ++t) {
    const int nt  = wave + t * 8;
    const int col = nt * 16 + nn;
    v8f z = {};
    v8f acc = wmma_tile<512>(A, f1T, nt * 16, lane, z);
    const float bb = b1[col];
#pragma unroll
    for (int r = 0; r < 8; ++r)
      Cb[(r + h * 8) * 256 + col] = acc[r] + bb;
  }
  __syncthreads();
  // per-row mean/var: wave w reduces rows 2w, 2w+1 with wave32 shuffles
  for (int rr = 0; rr < 2; ++rr) {
    const int r = wave * 2 + rr;
    float s = 0.f, s2 = 0.f;
    for (int c = lane; c < 256; c += 32) {
      const float x = Cb[r * 256 + c];
      s += x; s2 += x * x;
    }
#pragma unroll
    for (int off = 16; off > 0; off >>= 1) {
      s  += __shfl_down(s,  off);
      s2 += __shfl_down(s2, off);
    }
    if (lane == 0) {
      const float mu  = s * (1.f / 256.f);
      const float var = s2 * (1.f / 256.f) - mu * mu;
      mu_[r] = mu;
      rs_[r] = rsqrtf(var + 1e-5f);
    }
  }
  __syncthreads();
  for (int i = tid; i < 16 * 256; i += 256) {
    const int r = i >> 8, c = i & 255;
    const int row = row0 + r;
    if (row < MTOT) {
      const float x = (Cb[i] - mu_[r]) * rs_[r] * ln_g[c] + ln_b[c];
      h2[(size_t)row * 256 + c] = (bf16)fmaxf(x, 0.f);
    }
  }
}

// ---------------------------------------------------------------------------
// Kernel 4: h2 @ fus_w2 + b2 -> f32 output.
// ---------------------------------------------------------------------------
__global__ __launch_bounds__(256, 1)
void k_fus2(const bf16* __restrict__ h2,
            const bf16* __restrict__ f2T,
            const float* __restrict__ b2,
            float* __restrict__ out) {
  __shared__ bf16 A[16 * 256];
  const int row0 = blockIdx.x * 16;
  const int tid  = threadIdx.x;
  for (int i = tid; i < 16 * 256; i += 256) {
    const int r = i >> 8, k = i & 255;
    const int row = row0 + r;
    A[i] = (row < MTOT) ? h2[(size_t)row * 256 + k] : (bf16)0.f;
  }
  __syncthreads();
  const int wave = tid >> 5, lane = tid & 31;
  const int h = lane >> 4, nn = lane & 15;
#pragma clang loop unroll(disable)
  for (int t = 0; t < 2; ++t) {
    const int nt  = wave + t * 8;
    const int col = nt * 16 + nn;
    v8f z = {};
    v8f acc = wmma_tile<256>(A, f2T, nt * 16, lane, z);
    const float bb = b2[col];
#pragma unroll
    for (int r = 0; r < 8; ++r) {
      const int row = row0 + r + h * 8;
      if (row < MTOT)
        out[(size_t)row * 256 + col] = acc[r] + bb;
    }
  }
}

// ---------------------------------------------------------------------------
extern "C" void kernel_launch(void* const* d_in, const int* in_sizes, int n_in,
                              void* d_out, int out_size, void* d_ws, size_t ws_size,
                              hipStream_t stream) {
  const float* query   = (const float*)d_in[0];
  const float* refp    = (const float*)d_in[1];
  const float* img     = (const float*)d_in[2];
  const float* lidar   = (const float*)d_in[3];
  const float* l2i     = (const float*)d_in[4];
  const float* cam_w   = (const float*)d_in[5];
  const float* cam_b   = (const float*)d_in[6];
  const float* lidar_w = (const float*)d_in[7];
  const float* lidar_b = (const float*)d_in[8];
  const float* att_w1  = (const float*)d_in[9];
  const float* att_b1  = (const float*)d_in[10];
  const float* att_w2  = (const float*)d_in[11];
  const float* att_b2  = (const float*)d_in[12];
  const float* fus_w1  = (const float*)d_in[13];
  const float* fus_b1  = (const float*)d_in[14];
  const float* ln_g    = (const float*)d_in[15];
  const float* ln_b    = (const float*)d_in[16];
  const float* fus_w2  = (const float*)d_in[17];
  const float* fus_b2  = (const float*)d_in[18];
  float* out = (float*)d_out;
  (void)in_sizes; (void)n_in; (void)out_size; (void)ws_size;

  // workspace carve-out (256B aligned), ~7.2 MB total
  char* p = (char*)d_ws;
  auto take = [&](size_t n) { char* r = p; p += (n + 255) & ~(size_t)255; return r; };
  float* feat_cam = (float*)take((size_t)MTOT * 256 * 4);
  float* feat_lid = (float*)take((size_t)MTOT * 256 * 4);
  float* gate     = (float*)take((size_t)MTOT * 2 * 4);
  bf16*  cwT      = (bf16*)take((size_t)256 * 256 * 2);
  bf16*  lwT      = (bf16*)take((size_t)256 * 256 * 2);
  bf16*  f1T      = (bf16*)take((size_t)512 * 256 * 2);
  bf16*  f2T      = (bf16*)take((size_t)256 * 256 * 2);
  bf16*  fused    = (bf16*)take((size_t)MTOT * 512 * 2);
  bf16*  h2b      = (bf16*)take((size_t)MTOT * 256 * 2);

  hipLaunchKernelGGL(k_cvt_weights, dim3(512), dim3(256), 0, stream,
                     cam_w, lidar_w, fus_w1, fus_w2, cwT, lwT, f1T, f2T);
  hipLaunchKernelGGL(k_sample_gate, dim3(MTOT), dim3(256), 0, stream,
                     query, refp, img, lidar, l2i,
                     att_w1, att_b1, att_w2, att_b2,
                     feat_cam, feat_lid, gate);
  const int mblocks = (MTOT + 15) / 16;  // 113
  hipLaunchKernelGGL(k_proj_fuse, dim3(mblocks), dim3(256), 0, stream,
                     feat_cam, feat_lid, cwT, lwT, cam_b, lidar_b, gate, fused);
  hipLaunchKernelGGL(k_fus1_ln, dim3(mblocks), dim3(256), 0, stream,
                     fused, f1T, fus_b1, ln_g, ln_b, h2b);
  hipLaunchKernelGGL(k_fus2, dim3(mblocks), dim3(256), 0, stream,
                     h2b, f2T, fus_b2, out);
}
